// DGC_GRU_14645838479416
// MI455X (gfx1250) — matrix operand order repaired and checked by reading, not compiled, not run on hardware
//
#include <hip/hip_runtime.h>
#include <hip/hip_bf16.h>
#include <math.h>

// ---------------- problem constants ----------------
#define Bn    64
#define Nn    512
#define INF_  16      // IN_DIM
#define HISTn 24
#define FCn   48
#define TTn   (HISTn + FCn)   // 72
#define HIDn  64
#define ROWS  (Bn * Nn)       // 32768

typedef __attribute__((ext_vector_type(16))) _Float16 v16h;
typedef __attribute__((ext_vector_type(8)))  float    v8f;

// Assemble a v16h WMMA operand from two 16-byte chunks.
__device__ __forceinline__ v16h ld16h(const _Float16* p0, const _Float16* p1) {
    union { v16h v; uint4 q[2]; } u;
    u.q[0] = *(const uint4*)p0;
    u.q[1] = *(const uint4*)p1;
    return u.v;
}

__device__ __forceinline__ v8f wmma16(v16h a, v16h b, v8f c) {
    return __builtin_amdgcn_wmma_f32_16x16x32_f16(
        /*neg_a=*/false, a, /*neg_b=*/false, b,
        /*c_mod=*/(short)0, c, /*reuse_a=*/false, /*reuse_b=*/false);
}

__device__ __forceinline__ float sigm(float x) { return 1.0f / (1.0f + expf(-x)); }

// branchless wind gate: returns a if (v*c1 + u*c2 >= 0.5) else 0
__device__ __forceinline__ float gate(float u, float v, float c1v, float c2v, float a) {
    float w = fmaf(v, c1v, u * c2v);
    return (w >= 0.5f) ? a : 0.0f;   // unconditional 'a' -> v_cndmask, no branch
}

// ---------------- K0: one-time prep ----------------
__global__ void k0_cos(const float* __restrict__ angles, float* __restrict__ c1,
                       float* __restrict__ c2) {
    int g = blockIdx.x * blockDim.x + threadIdx.x;
    if (g < Nn * Nn) {
        float a = angles[g];
        c1[g] = sinf(a);          // cos(a - pi/2)
        c2[g] = cosf(a);
    }
}

__global__ void k0_weights(const float* __restrict__ W_ih, const float* __restrict__ W_hh,
                           _Float16* __restrict__ Wih16, _Float16* __restrict__ Whh16) {
    int g = blockIdx.x * blockDim.x + threadIdx.x;   // 192*64
    if (g >= 192 * 64) return;
    int n = g >> 6, k = g & 63;
    Whh16[n * 64 + k] = (_Float16)W_hh[n * 64 + k];
    if (k < 32) Wih16[n * 32 + k] = (k < 18) ? (_Float16)W_ih[n * 18 + k] : (_Float16)0.0f;
}

__global__ void k0_zeroh(_Float16* __restrict__ h16) {
    int g = blockIdx.x * blockDim.x + threadIdx.x;
    if (g < ROWS * HIDn) h16[g] = (_Float16)0.0f;
}

// ---------------- K1: wind gate degrees, dis, x / y staging ----------------
// one wave per (b,i) row; 8 waves per block; vectorized branch-free gate pass
__global__ void k1_deg(const float* __restrict__ feature, const float* __restrict__ pm25,
                       const float* __restrict__ adj, const float* __restrict__ c1,
                       const float* __restrict__ c2, const float* __restrict__ prev_out,
                       float* __restrict__ dis_ws, float* __restrict__ xbuf,
                       _Float16* __restrict__ yT, int t) {
    int lane = threadIdx.x & 31;
    int row  = blockIdx.x * 8 + (threadIdx.x >> 5);   // b*N + i
    int b = row >> 9, i = row & 511;

    int fbase = ((b * TTn + HISTn + t) * Nn + i) * INF_;
    float u = feature[fbase + 0];
    float v = feature[fbase + 1];

    const float4* r1 = (const float4*)(c1  + i * Nn);
    const float4* r2 = (const float4*)(c2  + i * Nn);
    const float4* ra = (const float4*)(adj + i * Nn);
    float deg = 0.0f;
    #pragma unroll
    for (int q = 0; q < 4; ++q) {                 // 128 float4 per row / 32 lanes
        int j4 = lane + q * 32;
        float4 a1 = r1[j4];
        float4 a2 = r2[j4];
        float4 aa = ra[j4];
        deg += gate(u, v, a1.x, a2.x, aa.x);
        deg += gate(u, v, a1.y, a2.y, aa.y);
        deg += gate(u, v, a1.z, a2.z, aa.z);
        deg += gate(u, v, a1.w, a2.w, aa.w);
    }
    #pragma unroll
    for (int off = 16; off > 0; off >>= 1) deg += __shfl_xor(deg, off, 32);
    float dis = (deg > 0.0f) ? rsqrtf(deg) : 0.0f;
    if (lane == 0) dis_ws[row] = dis;

    if (lane < 17) {
        float xv;
        if (lane == 0)
            xv = (t == 0) ? pm25[(b * HISTn + HISTn - 1) * Nn + i]
                          : prev_out[b * FCn * Nn + (t - 1) * Nn + i];
        else
            xv = feature[fbase + (lane - 1)];
        xbuf[row * 32 + lane] = xv;
        yT[b * (32 * Nn) + lane * Nn + i] = (_Float16)(dis * xv);
    } else {
        yT[b * (32 * Nn) + lane * Nn + i] = (_Float16)0.0f;
    }
}

// ---------------- K2: Tx1 = -dis_j * (A^T y) via WMMA ----------------
// one wave per (b, j-tile); branch-free vectorized mask build in LDS
__global__ void k2_tx1(const float* __restrict__ feature, const float* __restrict__ adj,
                       const float* __restrict__ c1, const float* __restrict__ c2,
                       const float* __restrict__ dis_ws, const _Float16* __restrict__ yT,
                       float* __restrict__ TxWS, int t) {
    __shared__ _Float16 At[16 * 32];     // [m=j_local][k=i_local]
    int lane = threadIdx.x;
    int wg = blockIdx.x;
    int b = wg >> 5;
    int j0 = (wg & 31) * 16;

    v8f acc0 = {};
    v8f acc1 = {};
    int m16 = lane & 15;
    int hi  = (lane < 16) ? 0 : 1;

    for (int i0 = 0; i0 < Nn; i0 += 32) {
        int i = i0 + lane;
        int fbase = ((b * TTn + HISTn + t) * Nn + i) * INF_;
        float u = feature[fbase + 0];
        float v = feature[fbase + 1];

        // 16 consecutive j's per lane, fetched as float4 (rows are 64B aligned)
        const float4* p1 = (const float4*)(c1  + i * Nn + j0);
        const float4* p2 = (const float4*)(c2  + i * Nn + j0);
        const float4* pa = (const float4*)(adj + i * Nn + j0);
        #pragma unroll
        for (int q = 0; q < 4; ++q) {
            float4 a1 = p1[q];
            float4 a2 = p2[q];
            float4 aa = pa[q];
            int jl = q * 4;
            At[(jl + 0) * 32 + lane] = (_Float16)gate(u, v, a1.x, a2.x, aa.x);
            At[(jl + 1) * 32 + lane] = (_Float16)gate(u, v, a1.y, a2.y, aa.y);
            At[(jl + 2) * 32 + lane] = (_Float16)gate(u, v, a1.z, a2.z, aa.z);
            At[(jl + 3) * 32 + lane] = (_Float16)gate(u, v, a1.w, a2.w, aa.w);
        }
        __syncthreads();

        const _Float16* ap = At + m16 * 32 + hi * 8;
        v16h a = ld16h(ap, ap + 16);

        const _Float16* yb = yT + b * (32 * Nn) + m16 * Nn + i0 + hi * 8;
        v16h b0 = ld16h(yb, yb + 16);
        v16h b1 = ld16h(yb + 16 * Nn, yb + 16 * Nn + 16);

        acc0 = wmma16(a, b0, acc0);
        acc1 = wmma16(a, b1, acc1);
        __syncthreads();
    }

    #pragma unroll
    for (int vg = 0; vg < 8; ++vg) {
        int m = vg + hi * 8;
        int row = b * Nn + j0 + m;
        float d = dis_ws[row];
        TxWS[row * 32 + m16]      = -d * acc0[vg];
        TxWS[row * 32 + 16 + m16] = -d * acc1[vg];
    }
}

// ---------------- K3: GCN sigmoid + xg (f16, padded 18->32) ----------------
__global__ void k3_gcn(const float* __restrict__ xbuf, const float* __restrict__ TxWS,
                       const float* __restrict__ W0, const float* __restrict__ W1,
                       const float* __restrict__ cheb_b, _Float16* __restrict__ xg) {
    int row = blockIdx.x * blockDim.x + threadIdx.x;
    if (row >= ROWS) return;
    float s = cheb_b[0];
    #pragma unroll
    for (int f = 0; f < 17; ++f)
        s += xbuf[row * 32 + f] * W0[f] + TxWS[row * 32 + f] * W1[f];
    float g = sigm(s);
    #pragma unroll
    for (int k = 0; k < 32; ++k) {
        float v = (k < 17) ? xbuf[row * 32 + k] : (k == 17 ? g : 0.0f);
        xg[row * 32 + k] = (_Float16)v;
    }
}

// ---------------- K4: fused GRU + output head via WMMA ----------------
// one wave per 16-row tile; 36 WMMAs per tile; gates in registers
__global__ void k4_gru(const _Float16* __restrict__ xg, const _Float16* __restrict__ Wih16,
                       const _Float16* __restrict__ Whh16, const float* __restrict__ b_ih,
                       const float* __restrict__ b_hh, const float* __restrict__ W_fc,
                       const float* __restrict__ b_fc, _Float16* __restrict__ h16,
                       float* __restrict__ out, int t) {
    __shared__ float ps[16];
    int lane = threadIdx.x;
    int r0 = blockIdx.x * 16;
    int m16 = lane & 15;
    int hi  = (lane < 16) ? 0 : 1;

    if (lane < 16) ps[lane] = 0.0f;
    __syncthreads();

    // A operands (constant across output tiles)
    const _Float16* xr = xg + (r0 + m16) * 32 + hi * 16;
    v16h a_x  = ld16h(xr, xr + 8);
    const _Float16* hr = h16 + (r0 + m16) * 64 + hi * 16;
    v16h a_h0 = ld16h(hr, hr + 8);          // K = 0..31
    v16h a_h1 = ld16h(hr + 32, hr + 40);    // K = 32..63

    float pacc[8];
    #pragma unroll
    for (int vg = 0; vg < 8; ++vg) pacc[vg] = 0.0f;

    for (int q = 0; q < 4; ++q) {
        // tiles: r-gate = q, z-gate = 4+q, n-gate = 8+q
        int nR = (q)     * 16 + m16;
        int nZ = (4 + q) * 16 + m16;
        int nN = (8 + q) * 16 + m16;

        const _Float16* pIR = Wih16 + nR * 32 + hi * 16;
        const _Float16* pIZ = Wih16 + nZ * 32 + hi * 16;
        const _Float16* pIN = Wih16 + nN * 32 + hi * 16;
        const _Float16* pHR = Whh16 + nR * 64 + hi * 16;
        const _Float16* pHZ = Whh16 + nZ * 64 + hi * 16;
        const _Float16* pHN = Whh16 + nN * 64 + hi * 16;

        v8f accR = {}, accZ = {}, accIN = {}, accHN = {};
        accR  = wmma16(a_x,  ld16h(pIR, pIR + 8),           accR);
        accR  = wmma16(a_h0, ld16h(pHR, pHR + 8),           accR);
        accR  = wmma16(a_h1, ld16h(pHR + 32, pHR + 40),     accR);
        accZ  = wmma16(a_x,  ld16h(pIZ, pIZ + 8),           accZ);
        accZ  = wmma16(a_h0, ld16h(pHZ, pHZ + 8),           accZ);
        accZ  = wmma16(a_h1, ld16h(pHZ + 32, pHZ + 40),     accZ);
        accIN = wmma16(a_x,  ld16h(pIN, pIN + 8),           accIN);
        accHN = wmma16(a_h0, ld16h(pHN, pHN + 8),           accHN);
        accHN = wmma16(a_h1, ld16h(pHN + 32, pHN + 40),     accHN);

        int c = m16 + 16 * q;
        float biR = b_ih[c],        bhR = b_hh[c];
        float biZ = b_ih[64 + c],   bhZ = b_hh[64 + c];
        float biN = b_ih[128 + c],  bhN = b_hh[128 + c];
        float wfc = W_fc[c];

        #pragma unroll
        for (int vg = 0; vg < 8; ++vg) {
            int row = r0 + vg + hi * 8;
            float r  = sigm(accR[vg] + biR + bhR);
            float z  = sigm(accZ[vg] + biZ + bhZ);
            float nn = tanhf(accIN[vg] + biN + r * (accHN[vg] + bhN));
            float hold = (float)h16[row * 64 + c];
            float hnew = (1.0f - z) * nn + z * hold;
            h16[row * 64 + c] = (_Float16)hnew;
            pacc[vg] += hnew * wfc;
        }
    }

    #pragma unroll
    for (int vg = 0; vg < 8; ++vg) atomicAdd(&ps[vg + hi * 8], pacc[vg]);
    __syncthreads();

    if (lane < 16) {
        int row = r0 + lane;
        int b = row >> 9, i = row & 511;
        out[b * FCn * Nn + t * Nn + i] = ps[lane] + b_fc[0];
    }
}

// ---------------- host launch ----------------
extern "C" void kernel_launch(void* const* d_in, const int* in_sizes, int n_in,
                              void* d_out, int out_size, void* d_ws, size_t ws_size,
                              hipStream_t stream) {
    const float* feature = (const float*)d_in[0];
    const float* pm25    = (const float*)d_in[1];
    const float* adj     = (const float*)d_in[2];
    const float* angles  = (const float*)d_in[3];
    const float* W0      = (const float*)d_in[4];
    const float* W1      = (const float*)d_in[5];
    const float* cheb_b  = (const float*)d_in[6];
    const float* W_ih    = (const float*)d_in[7];
    const float* W_hh    = (const float*)d_in[8];
    const float* b_ih    = (const float*)d_in[9];
    const float* b_hh    = (const float*)d_in[10];
    const float* W_fc    = (const float*)d_in[11];
    const float* b_fc    = (const float*)d_in[12];
    float* out = (float*)d_out;

    // workspace carve-up (256B aligned)
    char* w = (char*)d_ws;
    size_t off = 0;
    auto take = [&](size_t n) -> char* {
        char* p = w + off;
        off = (off + n + 255) & ~(size_t)255;
        return p;
    };
    float*    c1    = (float*)   take((size_t)Nn * Nn * 4);
    float*    c2    = (float*)   take((size_t)Nn * Nn * 4);
    float*    disw  = (float*)   take((size_t)ROWS * 4);
    float*    xbuf  = (float*)   take((size_t)ROWS * 32 * 4);
    _Float16* yT    = (_Float16*)take((size_t)Bn * 32 * Nn * 2);
    float*    TxWS  = (float*)   take((size_t)ROWS * 32 * 4);
    _Float16* xg    = (_Float16*)take((size_t)ROWS * 32 * 2);
    _Float16* h16   = (_Float16*)take((size_t)ROWS * HIDn * 2);
    _Float16* Wih16 = (_Float16*)take((size_t)192 * 32 * 2);
    _Float16* Whh16 = (_Float16*)take((size_t)192 * 64 * 2);
    (void)ws_size; (void)in_sizes; (void)n_in; (void)out_size;

    k0_cos    <<<(Nn * Nn + 255) / 256, 256, 0, stream>>>(angles, c1, c2);
    k0_weights<<<(192 * 64 + 255) / 256, 256, 0, stream>>>(W_ih, W_hh, Wih16, Whh16);
    k0_zeroh  <<<(ROWS * HIDn + 255) / 256, 256, 0, stream>>>(h16);

    for (int t = 0; t < FCn; ++t) {
        k1_deg<<<ROWS / 8, 256, 0, stream>>>(feature, pm25, adj, c1, c2, out,
                                             disw, xbuf, yT, t);
        k2_tx1<<<Bn * (Nn / 16), 32, 0, stream>>>(feature, adj, c1, c2, disw, yT,
                                                  TxWS, t);
        k3_gcn<<<(ROWS + 255) / 256, 256, 0, stream>>>(xbuf, TxWS, W0, W1, cheb_b, xg);
        k4_gru<<<ROWS / 16, 32, 0, stream>>>(xg, Wih16, Whh16, b_ih, b_hh, W_fc,
                                             b_fc, h16, out, t);
    }
}
